// SelfAttention_76957224010130
// MI455X (gfx1250) — matrix operand order repaired and checked by reading, not compiled, not run on hardware
//
#include <hip/hip_runtime.h>

#define N_SP 32768           // d*h*w
#define B_SZ 4
#define C_SZ 128

typedef __attribute__((ext_vector_type(16))) __bf16 bf16x16;
typedef __attribute__((ext_vector_type(8)))  __bf16 bf16x8;
typedef __attribute__((ext_vector_type(8)))  float  f32x8;

union BF16x16 { bf16x16 v; bf16x8 h[2]; __bf16 e[16]; };
union F32x8   { f32x8 v; float e[8]; };

// --- WMMA fragment loaders (CDNA5 16-bit layouts, wave32) -------------------
// A matrix 16(M)x32(K), source row-major [M][K], ld in elements.
// lanes 0-15: row=lane, h[0]=K0..7,  h[1]=K16..23
// lanes16-31: row=lane-16, h[0]=K8..15, h[1]=K24..31
__device__ __forceinline__ bf16x16 frag_a_rm(const __bf16* __restrict__ base, int ld) {
  const int lane = threadIdx.x & 31;
  const int r = lane & 15, hf = lane >> 4;
  const __bf16* p = base + (size_t)r * ld + hf * 8;
  BF16x16 u;
  u.h[0] = *(const bf16x8*)(p);
  u.h[1] = *(const bf16x8*)(p + 16);
  return u.v;
}
// B matrix 32(K)x16(N) sourced from Bt[N][K] row-major (K contiguous).
// lanes 0-15: col=lane, h[0]=K0..7, h[1]=K16..23 ; lanes16-31: col=lane-16, h[0]=K8..15, h[1]=K24..31
__device__ __forceinline__ bf16x16 frag_b_rmT(const __bf16* __restrict__ base, int ld) {
  const int lane = threadIdx.x & 31;
  const int r = lane & 15, hf = lane >> 4;
  const __bf16* p = base + (size_t)r * ld + hf * 8;
  BF16x16 u;
  u.h[0] = *(const bf16x8*)(p);
  u.h[1] = *(const bf16x8*)(p + 16);
  return u.v;
}
__device__ __forceinline__ f32x8 wmma_bf16(bf16x16 a, bf16x16 b, f32x8 c) {
  return __builtin_amdgcn_wmma_f32_16x16x32_bf16(false, a, false, b, (short)0, c,
                                                 false, false);
}

// ---------------------------------------------------------------------------
// K1a: GroupNorm partial sums. grid = 16 (b,g) * 64 partials, block 256.
__global__ void k_gnsum(const float* __restrict__ inp, float* __restrict__ partials) {
  __shared__ float s1[256], s2[256];
  const int bg = blockIdx.x >> 6;
  const int part = blockIdx.x & 63;
  const size_t base = (size_t)bg * (32 * (size_t)N_SP) + (size_t)part * 16384;
  const int t = threadIdx.x;
  float s = 0.f, q = 0.f;
  for (int i = t; i < 16384; i += 256) { float v = inp[base + i]; s += v; q += v * v; }
  s1[t] = s; s2[t] = q; __syncthreads();
  for (int ofs = 128; ofs > 0; ofs >>= 1) {
    if (t < ofs) { s1[t] += s1[t + ofs]; s2[t] += s2[t + ofs]; }
    __syncthreads();
  }
  if (t == 0) { partials[blockIdx.x * 2] = s1[0]; partials[blockIdx.x * 2 + 1] = s2[0]; }
}

// K1b: finalize mean/rstd. grid = 16, block 64.
__global__ void k_gnstat(const float* __restrict__ partials, float* __restrict__ stats) {
  __shared__ float s1[64], s2[64];
  const int bg = blockIdx.x, t = threadIdx.x;
  s1[t] = partials[(bg * 64 + t) * 2];
  s2[t] = partials[(bg * 64 + t) * 2 + 1];
  __syncthreads();
  for (int ofs = 32; ofs > 0; ofs >>= 1) {
    if (t < ofs) { s1[t] += s1[t + ofs]; s2[t] += s2[t + ofs]; }
    __syncthreads();
  }
  if (t == 0) {
    const float N = 32.f * (float)N_SP;
    float mean = s1[0] / N;
    float var = s2[0] / N - mean * mean;
    stats[bg * 2] = mean;
    stats[bg * 2 + 1] = rsqrtf(var + 1e-5f);
  }
}

// K2: weights f32 -> bf16. grid 192, block 256.
__global__ void k_wconv(const float* __restrict__ wqkv, const float* __restrict__ wout,
                        __bf16* __restrict__ wqkvbf, __bf16* __restrict__ woutbf) {
  int i = blockIdx.x * 256 + threadIdx.x;
  if (i < 384 * 128) wqkvbf[i] = (__bf16)wqkv[i];
  if (i < 128 * 128) woutbf[i] = (__bf16)wout[i];
}

// K3: normalize + transpose -> normT[b][n][c] bf16. grid = b*4cg*512, block 256.
__global__ void k_normT(const float* __restrict__ inp, const float* __restrict__ gamma,
                        const float* __restrict__ beta, const float* __restrict__ stats,
                        __bf16* __restrict__ normT) {
  __shared__ __bf16 tile[32][72];
  int id = blockIdx.x;
  const int nblk = id & 511; id >>= 9;
  const int cg = id & 3;  const int b = id >> 2;
  const int c0 = cg * 32, n0 = nblk * 64;
  const int t = threadIdx.x;
  const float mu = stats[(b * 4 + cg) * 2];
  const float rs = stats[(b * 4 + cg) * 2 + 1];
  for (int i = t; i < 2048; i += 256) {
    int c = i >> 6, nn = i & 63;
    float v = inp[((size_t)(b * C_SZ + c0 + c)) * N_SP + n0 + nn];
    float y = (v - mu) * rs * gamma[c0 + c] + beta[c0 + c];
    tile[c][nn] = (__bf16)y;
  }
  __syncthreads();
  for (int i = t; i < 2048; i += 256) {
    int nn = i >> 5, c = i & 31;
    normT[((size_t)b * N_SP + n0 + nn) * C_SZ + c0 + c] = tile[c][nn];
  }
}

// K4: QKV GEMM: qkv[b][o][n] = Wqkv[o][c] * norm[b][c][n]. grid 2048, block 256.
// 8 waves; each wave: 3 m-tiles x 4 n-tiles (96 acc VGPRs) -> needs the
// 2-waves/SIMD register budget (256 VGPRs) to stay spill-free.
__global__ void __launch_bounds__(256, 2)
k_qkv(const __bf16* __restrict__ wbf, const __bf16* __restrict__ normT,
      __bf16* __restrict__ qkv) {
  __shared__ __attribute__((aligned(16))) __bf16 Nt[64][136];   // 16B row pad
  const int bb = blockIdx.x >> 9;
  const int n0 = (blockIdx.x & 511) * 64;
  const int t = threadIdx.x;
  const __bf16* nsrc = normT + ((size_t)bb * N_SP + n0) * C_SZ;
  for (int i = t; i < 64 * 16; i += 256) {
    int row = i >> 4, c8 = (i & 15) * 8;
    *(bf16x8*)(&Nt[row][c8]) = *(const bf16x8*)(nsrc + (size_t)row * C_SZ + c8);
  }
  __syncthreads();
  const int wv = t >> 5;
  f32x8 acc[3][4] = {};
#pragma unroll
  for (int ks = 0; ks < 4; ++ks) {
    const int k0 = ks * 32;
    bf16x16 bfr[4];
#pragma unroll
    for (int nt = 0; nt < 4; ++nt)
      bfr[nt] = frag_b_rmT(&Nt[nt * 16][k0], 136);
#pragma unroll
    for (int mi = 0; mi < 3; ++mi) {
      const int m0 = (wv + mi * 8) * 16;
      bf16x16 af = frag_a_rm(wbf + (size_t)m0 * 128 + k0, 128);
#pragma unroll
      for (int nt = 0; nt < 4; ++nt)
        acc[mi][nt] = wmma_bf16(af, bfr[nt], acc[mi][nt]);
    }
  }
  const int lane = t & 31, col = lane & 15, hf = lane >> 4;
#pragma unroll
  for (int mi = 0; mi < 3; ++mi) {
    const int m0 = (wv + mi * 8) * 16;
#pragma unroll
    for (int nt = 0; nt < 4; ++nt) {
      F32x8 u; u.v = acc[mi][nt];
#pragma unroll
      for (int r = 0; r < 8; ++r) {
        int o = m0 + r + 8 * hf;
        int n = n0 + nt * 16 + col;
        qkv[((size_t)(bb * 384 + o)) * N_SP + n] = (__bf16)u.e[r];
      }
    }
  }
}

// K5: softmax over n of k rows. grid 512 (b*128), block 256.
__global__ void k_softmax(const __bf16* __restrict__ qkv, __bf16* __restrict__ ksoft) {
  __shared__ float red[256];
  const int row = blockIdx.x;
  const int b = row >> 7, o = row & 127;
  const __bf16* src = qkv + ((size_t)(b * 384 + 128 + o)) * N_SP;
  __bf16* dst = ksoft + ((size_t)(b * C_SZ + o)) * N_SP;
  const int t = threadIdx.x;
  float m = -3.4e38f;
  for (int i = t; i < N_SP; i += 256) m = fmaxf(m, (float)src[i]);
  red[t] = m; __syncthreads();
  for (int ofs = 128; ofs > 0; ofs >>= 1) {
    if (t < ofs) red[t] = fmaxf(red[t], red[t + ofs]);
    __syncthreads();
  }
  m = red[0]; __syncthreads();
  float s = 0.f;
  for (int i = t; i < N_SP; i += 256) s += __expf((float)src[i] - m);
  red[t] = s; __syncthreads();
  for (int ofs = 128; ofs > 0; ofs >>= 1) {
    if (t < ofs) red[t] += red[t + ofs];
    __syncthreads();
  }
  const float invs = 1.0f / red[0];
  for (int i = t; i < N_SP; i += 256)
    dst[i] = (__bf16)(__expf((float)src[i] - m) * invs);
}

// K6: transpose q -> qT[bh][n][d]. grid b*h*512, block 256.
__global__ void k_qT(const __bf16* __restrict__ qkv, __bf16* __restrict__ qT) {
  __shared__ __bf16 tile[32][72];
  int id = blockIdx.x;
  const int nblk = id & 511; id >>= 9;
  const int h = id & 3; const int b = id >> 2;
  const int n0 = nblk * 64;
  const int t = threadIdx.x;
  for (int i = t; i < 2048; i += 256) {
    int d = i >> 6, nn = i & 63;
    tile[d][nn] = qkv[((size_t)(b * 384 + h * 32 + d)) * N_SP + n0 + nn];
  }
  __syncthreads();
  for (int i = t; i < 2048; i += 256) {
    int nn = i >> 5, d = i & 31;
    qT[(((size_t)(b * 4 + h)) * N_SP + n0 + nn) * 32 + d] = tile[d][nn];
  }
}

// K7: context partials: ctx[d][e] += sum_n k[d][n]*v[e][n], 32-way K split.
// grid 256, block 256 (2048 waves = 16 bh * 4 tiles * 32 ksplits).
__global__ void __launch_bounds__(256, 2)
k_ctx(const __bf16* __restrict__ ksoft, const __bf16* __restrict__ qkv,
      float* __restrict__ ctxpart) {
  const int w = blockIdx.x * 8 + (threadIdx.x >> 5);
  const int bh = w >> 7;
  const int rem = w & 127;
  const int tile = rem >> 5, ksplit = rem & 31;
  const int b = bh >> 2, h = bh & 3;
  const int d0 = (tile >> 1) * 16, e0 = (tile & 1) * 16;
  const __bf16* abase = ksoft + ((size_t)(bh * 32 + d0)) * N_SP;
  const __bf16* bbase = qkv + ((size_t)(b * 384 + 256 + h * 32 + e0)) * N_SP;
  const int nstart = ksplit * 1024;
  f32x8 acc = {};
#pragma unroll 4
  for (int kk = 0; kk < 32; ++kk) {
    const int k0 = nstart + kk * 32;
    if ((kk & 7) == 0) {
      __builtin_prefetch(abase + k0 + 512, 0, 0);
      __builtin_prefetch(bbase + k0 + 512, 0, 0);
    }
    acc = wmma_bf16(frag_a_rm(abase + k0, N_SP), frag_b_rmT(bbase + k0, N_SP), acc);
  }
  float* dst = ctxpart + ((size_t)((ksplit * 16 + bh) * 4 + tile)) * 256;
  const int lane = threadIdx.x & 31, col = lane & 15, hf = lane >> 4;
  F32x8 u; u.v = acc;
#pragma unroll
  for (int r = 0; r < 8; ++r) dst[(r + 8 * hf) * 16 + col] = u.e[r];
}

// K8: reduce ksplits -> ctx[bh][d][e] f32. grid 16, block 256.
__global__ void k_ctxred(const float* __restrict__ ctxpart, float* __restrict__ ctx) {
  const int bh = blockIdx.x, t = threadIdx.x;
  for (int i = t; i < 1024; i += 256) {
    const int tile = i >> 8, idx = i & 255;
    float s = 0.f;
    for (int ks = 0; ks < 32; ++ks)
      s += ctxpart[((size_t)((ks * 16 + bh) * 4 + tile)) * 256 + idx];
    const int dr = (tile >> 1) * 16 + (idx >> 4);
    const int ec = (tile & 1) * 16 + (idx & 15);
    ctx[bh * 1024 + dr * 32 + ec] = s;
  }
}

// A-frag for out GEMM: A[e][d] = ctx[d][e] (f32 gather -> bf16).
__device__ __forceinline__ bf16x16 frag_a_ctxT(const float* __restrict__ ctx, int e0) {
  const int lane = threadIdx.x & 31;
  const int r = lane & 15, hf = lane >> 4;
  const int e = e0 + r;
  BF16x16 u;
#pragma unroll
  for (int i = 0; i < 8; ++i) {
    u.e[i]     = (__bf16)ctx[(hf * 8 + i) * 32 + e];
    u.e[8 + i] = (__bf16)ctx[(16 + hf * 8 + i) * 32 + e];
  }
  return u.v;
}

// K9: out[e][n] = sum_d ctx[d][e]*q[d][n]; write transposed attnT[b][n][c] bf16.
// grid 256, block 256 (2048 waves = 16 bh * 128 n-groups of 256).
__global__ void __launch_bounds__(256, 2)
k_attn(const float* __restrict__ ctx, const __bf16* __restrict__ qT,
       __bf16* __restrict__ attnT) {
  __shared__ __bf16 lds[8][32][24];
  const int wid = threadIdx.x >> 5;
  const int w = blockIdx.x * 8 + wid;
  const int bh = w >> 7, ng = w & 127;
  const int b = bh >> 2, h = bh & 3;
  const int nbase = ng * 256;
  const bf16x16 a0 = frag_a_ctxT(ctx + bh * 1024, 0);
  const bf16x16 a1 = frag_a_ctxT(ctx + bh * 1024, 16);
  const int lane = threadIdx.x & 31, col = lane & 15, hf = lane >> 4;
  for (int tI = 0; tI < 16; ++tI) {
    const int n0 = nbase + tI * 16;
    bf16x16 bfr = frag_b_rmT(qT + ((size_t)bh * N_SP + n0) * 32, 32);
    f32x8 z = {};
    f32x8 c0 = wmma_bf16(a0, bfr, z);
    f32x8 c1 = wmma_bf16(a1, bfr, z);
    F32x8 u0, u1; u0.v = c0; u1.v = c1;
#pragma unroll
    for (int r = 0; r < 8; ++r) {
      lds[wid][r + 8 * hf][col]      = (__bf16)u0.e[r];
      lds[wid][16 + r + 8 * hf][col] = (__bf16)u1.e[r];
    }
    __syncthreads();
#pragma unroll
    for (int j = 0; j < 16; ++j) {
      const size_t n = (size_t)n0 + j;
      attnT[((size_t)b * N_SP + n) * C_SZ + h * 32 + lane] = lds[wid][lane][j];
    }
    __syncthreads();
  }
}

// K10: out-proj + bias + residual -> f32 output [b][c][n]. grid 2048, block 256.
__global__ void __launch_bounds__(256, 2)
k_outproj(const __bf16* __restrict__ woutbf, const __bf16* __restrict__ attnT,
          const float* __restrict__ bias, const float* __restrict__ inp,
          float* __restrict__ out) {
  __shared__ __attribute__((aligned(16))) __bf16 Nt[64][136];
  const int bb = blockIdx.x >> 9;
  const int n0 = (blockIdx.x & 511) * 64;
  const int t = threadIdx.x;
  const __bf16* nsrc = attnT + ((size_t)bb * N_SP + n0) * C_SZ;
  for (int i = t; i < 64 * 16; i += 256) {
    int row = i >> 4, c8 = (i & 15) * 8;
    *(bf16x8*)(&Nt[row][c8]) = *(const bf16x8*)(nsrc + (size_t)row * C_SZ + c8);
  }
  __syncthreads();
  const int wv = t >> 5, lane = t & 31, col = lane & 15, hf = lane >> 4;
  const int m0 = wv * 16;
  f32x8 acc[4] = {};
#pragma unroll
  for (int ks = 0; ks < 4; ++ks) {
    const int k0 = ks * 32;
    bf16x16 af = frag_a_rm(woutbf + (size_t)m0 * 128 + k0, 128);
#pragma unroll
    for (int nt = 0; nt < 4; ++nt)
      acc[nt] = wmma_bf16(af, frag_b_rmT(&Nt[nt * 16][k0], 136), acc[nt]);
  }
#pragma unroll
  for (int nt = 0; nt < 4; ++nt) {
    F32x8 u; u.v = acc[nt];
#pragma unroll
    for (int r = 0; r < 8; ++r) {
      const int o = m0 + r + 8 * hf;
      const size_t idx = ((size_t)(bb * C_SZ + o)) * N_SP + n0 + nt * 16 + col;
      out[idx] = u.e[r] + bias[o] + inp[idx];
    }
  }
}

// ---------------------------------------------------------------------------
extern "C" void kernel_launch(void* const* d_in, const int* in_sizes, int n_in,
                              void* d_out, int out_size, void* d_ws, size_t ws_size,
                              hipStream_t stream) {
  (void)in_sizes; (void)n_in; (void)out_size; (void)ws_size;
  const float* inp  = (const float*)d_in[0];
  const float* gnw  = (const float*)d_in[1];
  const float* gnb  = (const float*)d_in[2];
  const float* wqkv = (const float*)d_in[3];
  const float* wout = (const float*)d_in[4];
  const float* outb = (const float*)d_in[5];
  float* out = (float*)d_out;

  char* ws = (char*)d_ws;
  size_t cur = 0;
  auto alloc = [&](size_t bytes) -> void* {
    void* p = ws + cur;
    cur = (cur + bytes + 255) & ~(size_t)255;
    return p;
  };
  float*  partials = (float*)alloc(1024 * 2 * sizeof(float));
  float*  stats    = (float*)alloc(16 * 2 * sizeof(float));
  __bf16* wqkvbf   = (__bf16*)alloc((size_t)384 * 128 * 2);
  __bf16* woutbf   = (__bf16*)alloc((size_t)128 * 128 * 2);
  __bf16* normT    = (__bf16*)alloc((size_t)B_SZ * N_SP * C_SZ * 2);
  __bf16* qkvb     = (__bf16*)alloc((size_t)B_SZ * 384 * N_SP * 2);
  __bf16* ksoft    = (__bf16*)alloc((size_t)B_SZ * C_SZ * N_SP * 2);
  __bf16* qT       = (__bf16*)alloc((size_t)B_SZ * C_SZ * N_SP * 2);
  float*  ctxpart  = (float*)alloc((size_t)2048 * 256 * sizeof(float));
  float*  ctx      = (float*)alloc((size_t)16 * 1024 * sizeof(float));
  __bf16* attnT    = (__bf16*)alloc((size_t)B_SZ * N_SP * C_SZ * 2);

  k_gnsum  <<<1024, 256, 0, stream>>>(inp, partials);
  k_gnstat <<<16,   64,  0, stream>>>(partials, stats);
  k_wconv  <<<192,  256, 0, stream>>>(wqkv, wout, wqkvbf, woutbf);
  k_normT  <<<8192, 256, 0, stream>>>(inp, gnw, gnb, stats, normT);
  k_qkv    <<<2048, 256, 0, stream>>>(wqkvbf, normT, qkvb);
  k_softmax<<<512,  256, 0, stream>>>(qkvb, ksoft);
  k_qT     <<<8192, 256, 0, stream>>>(qkvb, qT);
  k_ctx    <<<256,  256, 0, stream>>>(ksoft, qkvb, ctxpart);
  k_ctxred <<<16,   256, 0, stream>>>(ctxpart, ctx);
  k_attn   <<<256,  256, 0, stream>>>(ctx, qT, attnT);
  k_outproj<<<2048, 256, 0, stream>>>(woutbf, attnT, outb, inp, out);
}
